// GCN_79405355369095
// MI455X (gfx1250) — matrix operand order repaired and checked by reading, not compiled, not run on hardware
//
#include <hip/hip_runtime.h>
#include <hip/hip_bf16.h>

typedef __attribute__((ext_vector_type(16))) _Float16 v16h;
typedef __attribute__((ext_vector_type(8)))  _Float16 v8h;
typedef __attribute__((ext_vector_type(8)))  float    v8f;

#define N_NODES 100000
#define N_EDGES 1600000

// ---------------- workspace layout (bytes) ----------------
#define XH_OFF   ((size_t)0)            // x f16 [100000*128], later reused as z2h f16 [100000*64]
#define H_OFF    ((size_t)25600000)     // h fp32 [100000*128] (h1, then h2 in first half)
#define AGG_OFF  ((size_t)76800000)     // agg fp32 [100000*128]
#define ZH_OFF   ((size_t)128000000)    // z f16 [100000*128]
#define DEG_OFF  ((size_t)153600000)    // deg fp32 [100000]
#define ISD_OFF  ((size_t)154000000)    // deg^-1/2
#define IND_OFF  ((size_t)154400000)    // deg^-1
#define W1T_OFF  ((size_t)154800000)    // W1^T f16 [128*128]
#define W2T_OFF  ((size_t)154832768)    // W2^T f16 [64*128]
#define FW1T_OFF ((size_t)154849152)    // fw1^T f16 [64*128]
#define FW2T_OFF ((size_t)154865536)    // fw2^T f16 [32*64]
#define FW3T_OFF ((size_t)154869632)    // fw3^T f16 [16*32]

// ---------------- fragment builders (CDNA5 WMMA layouts, vectorized loads) ----------------
// A-matrix 16x32 f16: lane holds row (lane&15); element i -> K = hi*8+i (i<8),
// 16+hi*8+(i-8) (i>=8). Two contiguous 16-byte runs per lane -> 2x b128 loads.
template <typename P>
__device__ __forceinline__ v16h fragA_vec(P rowp, int k0, int hi) {
  v8h lo = *(const v8h*)(&rowp[k0 + hi * 8]);
  v8h hs = *(const v8h*)(&rowp[k0 + 16 + hi * 8]);
  v16h f;
#pragma unroll
  for (int i = 0; i < 8; ++i) { f[i] = lo[i]; f[8 + i] = hs[i]; }
  return f;
}

// B-matrix 32x16 f16 from TRANSPOSED weights Wt[N x K] row-major: lane holds
// col (lane&15); element i -> K = hi*8+i (i<8), 16+hi*8+(i-8) (i>=8). Two 16-byte runs.
__device__ __forceinline__ v16h fragBT(const _Float16* __restrict__ Bt, int ldk, int k0,
                                       int col, int hi) {
  const _Float16* rp = Bt + (size_t)col * ldk + k0 + hi * 8;
  v8h lo = *(const v8h*)rp;
  v8h hs = *(const v8h*)(rp + 16);
  v16h f;
#pragma unroll
  for (int i = 0; i < 8; ++i) { f[i] = lo[i]; f[8 + i] = hs[i]; }
  return f;
}

__device__ __forceinline__ void wave_lds_sync() {
  __builtin_amdgcn_wave_barrier();
  asm volatile("s_wait_dscnt 0x0" ::: "memory");
  __builtin_amdgcn_wave_barrier();
}

// ---------------- utility kernels ----------------
__global__ void k_f32_to_f16(const float* __restrict__ in, _Float16* __restrict__ out, int n) {
  int i = blockIdx.x * blockDim.x + threadIdx.x;
  if (i < n) out[i] = (_Float16)in[i];
}

// out[n*K + k] = (f16) in[k*N + n]   (weights [K x N] -> transposed f16 [N x K])
__global__ void k_transpose_f16(const float* __restrict__ in, _Float16* __restrict__ out,
                                int K, int N) {
  int i = blockIdx.x * blockDim.x + threadIdx.x;
  if (i < K * N) {
    int k = i / N, n = i - k * N;
    out[(size_t)n * K + k] = (_Float16)in[i];
  }
}

__global__ void k_zero_f32(float* __restrict__ p, int n) {
  int i = blockIdx.x * blockDim.x + threadIdx.x;
  if (i < n) p[i] = 0.0f;
}

__global__ void k_deg_init(float* __restrict__ deg, int n) {
  int i = blockIdx.x * blockDim.x + threadIdx.x;
  if (i < n) deg[i] = 1.0f;  // self-loop
}

__global__ void k_deg_count(const int* __restrict__ dst, float* __restrict__ deg, int nE) {
  int e = blockIdx.x * blockDim.x + threadIdx.x;
  if (e < nE) atomicAdd(&deg[dst[e]], 1.0f);
}

__global__ void k_deg_norm(const float* __restrict__ deg, float* __restrict__ isd,
                           float* __restrict__ ind, int n) {
  int i = blockIdx.x * blockDim.x + threadIdx.x;
  if (i < n) {
    float d = deg[i];
    isd[i] = rsqrtf(d);
    ind[i] = 1.0f / d;
  }
}

// ---------------- node-feature GEMM: C[M,N] = A[M,K](f16) @ Bt[N,K](f16)^T ----------------
__global__ void k_gemm(const _Float16* __restrict__ A, const _Float16* __restrict__ Bt,
                       float* __restrict__ C, int M, int K, int N) {
  int lane = threadIdx.x & 31;
  int wv = threadIdx.x >> 5;
  int tiles_n = N >> 4;
  int tile = blockIdx.x * (blockDim.x >> 5) + wv;
  if (tile >= (M >> 4) * tiles_n) return;  // wave-uniform
  int tm = tile / tiles_n, tn = tile - tm * tiles_n;
  int n15 = lane & 15, hi = lane >> 4;
  const _Float16* arow = A + (size_t)(tm * 16 + n15) * K;
  v8f acc = {};
  for (int k0 = 0; k0 < K; k0 += 32) {
    v16h a = fragA_vec(arow, k0, hi);
    v16h b = fragBT(Bt, K, k0, tn * 16 + n15, hi);
    acc = __builtin_amdgcn_wmma_f32_16x16x32_f16(false, a, false, b, (short)0, acc, false, false);
  }
#pragma unroll
  for (int v = 0; v < 8; ++v)
    C[(size_t)(tm * 16 + v + 8 * hi) * N + tn * 16 + n15] = acc[v];
}

// ---------------- edge scatter-aggregate (L2-resident fp32 atomics) ----------------
template <int CH>
__global__ void k_agg(const int* __restrict__ src, const int* __restrict__ dst,
                      const float* __restrict__ isd, const float* __restrict__ h,
                      float* __restrict__ agg, int nE) {
  long idx = (long)blockIdx.x * blockDim.x + threadIdx.x;
  int e = (int)(idx >> 5);
  int lane = (int)(idx & 31);
  if (e >= nE) return;
  int s = src[e], d = dst[e];
  float coef = isd[s] * isd[d];
  const int cpl = CH / 32;
  const float* hp = h + (size_t)s * CH + lane * cpl;
  float* ap = agg + (size_t)d * CH + lane * cpl;
#pragma unroll
  for (int j = 0; j < cpl; ++j) atomicAdd(&ap[j], hp[j] * coef);
}

// z = [relu](agg + h*inv_deg + bias), written as f16 for the next WMMA stage
template <int CH, bool RELU>
__global__ void k_finalize(const float* __restrict__ agg, const float* __restrict__ h,
                           const float* __restrict__ ind, const float* __restrict__ b,
                           _Float16* __restrict__ outh, int nNodes) {
  long i = (long)blockIdx.x * blockDim.x + threadIdx.x;
  if (i >= (long)nNodes * CH) return;
  int node = (int)(i / CH), c = (int)(i % CH);
  float v = agg[i] + h[i] * ind[node] + b[c];
  if (RELU) v = fmaxf(v, 0.0f);
  outh[i] = (_Float16)v;
}

// ---------------- per-edge decoder MLP: 128 -> 64 -> 32 -> 16 -> 1 ----------------
// One wave handles a tile of 16 edges; all dense layers via v_wmma_f32_16x16x32_f16,
// D->A re-layout through per-wave LDS (s_wait_dscnt sync, no block barrier).
#define LDW 72  // padded f16 row stride in LDS (144 B: 16B-aligned, bank-spread)
__global__ void __launch_bounds__(256) k_decode(
    const int* __restrict__ src, const int* __restrict__ dst,
    const _Float16* __restrict__ z2h,
    const _Float16* __restrict__ fw1t, const float* __restrict__ fb1,
    const _Float16* __restrict__ fw2t, const float* __restrict__ fb2,
    const _Float16* __restrict__ fw3t, const float* __restrict__ fb3,
    const float* __restrict__ fw4, const float* __restrict__ fb4,
    float* __restrict__ out, int nTiles) {
  __shared__ __align__(16) _Float16 smem[8 * 16 * LDW];
  int lane = threadIdx.x & 31;
  int wv = threadIdx.x >> 5;
  int tile = blockIdx.x * 8 + wv;
  if (tile >= nTiles) return;  // wave-uniform
  int n15 = lane & 15, hi = lane >> 4;

  _Float16* smw = smem + wv * 16 * LDW;           // this wave's 16-row tile
  _Float16* rowp = smw + n15 * LDW;               // this lane's A-row in LDS
  float* s3 = (float*)smw;                        // f32 alias for layer-3 output

  int e = tile * 16 + n15;
  int s = src[e], d = dst[e];

  // A fragments for feat = concat(z2[src](64), z2[dst](64)), K = 128
  v16h a[4];
  a[0] = fragA_vec(z2h + (size_t)s * 64, 0, hi);
  a[1] = fragA_vec(z2h + (size_t)s * 64, 32, hi);
  a[2] = fragA_vec(z2h + (size_t)d * 64, 0, hi);
  a[3] = fragA_vec(z2h + (size_t)d * 64, 32, hi);

  // ---- layer 1: 128 -> 64, relu ----
#pragma unroll
  for (int nt = 0; nt < 4; ++nt) {
    v8f acc = {};
#pragma unroll
    for (int kt = 0; kt < 4; ++kt) {
      v16h b = fragBT(fw1t, 128, kt * 32, nt * 16 + n15, hi);
      acc = __builtin_amdgcn_wmma_f32_16x16x32_f16(false, a[kt], false, b, (short)0, acc, false, false);
    }
    float bias = fb1[nt * 16 + n15];
#pragma unroll
    for (int v = 0; v < 8; ++v) {
      float xv = fmaxf(acc[v] + bias, 0.0f);
      smw[(v + 8 * hi) * LDW + nt * 16 + n15] = (_Float16)xv;
    }
  }
  wave_lds_sync();

  // ---- layer 2: 64 -> 32, relu ----
  v16h a2[2];
  a2[0] = fragA_vec(rowp, 0, hi);
  a2[1] = fragA_vec(rowp, 32, hi);
  v8f acc2[2];
#pragma unroll
  for (int nt = 0; nt < 2; ++nt) {
    acc2[nt] = (v8f){};
#pragma unroll
    for (int kt = 0; kt < 2; ++kt) {
      v16h b = fragBT(fw2t, 64, kt * 32, nt * 16 + n15, hi);
      acc2[nt] = __builtin_amdgcn_wmma_f32_16x16x32_f16(false, a2[kt], false, b, (short)0, acc2[nt], false, false);
    }
  }
  wave_lds_sync();  // LDS reads done before overwrite
#pragma unroll
  for (int nt = 0; nt < 2; ++nt) {
    float bias = fb2[nt * 16 + n15];
#pragma unroll
    for (int v = 0; v < 8; ++v) {
      float xv = fmaxf(acc2[nt][v] + bias, 0.0f);
      smw[(v + 8 * hi) * LDW + nt * 16 + n15] = (_Float16)xv;
    }
  }
  wave_lds_sync();

  // ---- layer 3: 32 -> 16, relu ----
  v16h a3 = fragA_vec(rowp, 0, hi);
  v16h b3 = fragBT(fw3t, 32, 0, n15, hi);
  v8f acc3 = {};
  acc3 = __builtin_amdgcn_wmma_f32_16x16x32_f16(false, a3, false, b3, (short)0, acc3, false, false);
  wave_lds_sync();
  {
    float bias = fb3[n15];
#pragma unroll
    for (int v = 0; v < 8; ++v)
      s3[(v + 8 * hi) * 18 + n15] = fmaxf(acc3[v] + bias, 0.0f);
  }
  wave_lds_sync();

  // ---- layer 4: 16 -> 1 (scalar dot, fp32) ----
  if (lane < 16) {
    float sum = fb4[0];
#pragma unroll
    for (int j = 0; j < 16; ++j) sum += s3[lane * 18 + j] * fw4[j];
    out[tile * 16 + lane] = sum;
  }
}

// ---------------- orchestration ----------------
extern "C" void kernel_launch(void* const* d_in, const int* in_sizes, int n_in,
                              void* d_out, int out_size, void* d_ws, size_t ws_size,
                              hipStream_t stream) {
  const float* x   = (const float*)d_in[0];
  const int*   ei  = (const int*)d_in[1];
  const float* W1  = (const float*)d_in[2];
  const float* b1  = (const float*)d_in[3];
  const float* W2  = (const float*)d_in[4];
  const float* b2  = (const float*)d_in[5];
  const float* fw1 = (const float*)d_in[6];
  const float* fb1 = (const float*)d_in[7];
  const float* fw2 = (const float*)d_in[8];
  const float* fb2 = (const float*)d_in[9];
  const float* fw3 = (const float*)d_in[10];
  const float* fb3 = (const float*)d_in[11];
  const float* fw4 = (const float*)d_in[12];
  const float* fb4 = (const float*)d_in[13];
  float* out = (float*)d_out;

  const int* src = ei;
  const int* dst = ei + N_EDGES;

  char* ws = (char*)d_ws;
  _Float16* xh   = (_Float16*)(ws + XH_OFF);   // then z2h
  float*    h    = (float*)(ws + H_OFF);
  float*    agg  = (float*)(ws + AGG_OFF);
  _Float16* zh   = (_Float16*)(ws + ZH_OFF);
  float*    deg  = (float*)(ws + DEG_OFF);
  float*    isd  = (float*)(ws + ISD_OFF);
  float*    ind  = (float*)(ws + IND_OFF);
  _Float16* w1t  = (_Float16*)(ws + W1T_OFF);
  _Float16* w2t  = (_Float16*)(ws + W2T_OFF);
  _Float16* fw1t = (_Float16*)(ws + FW1T_OFF);
  _Float16* fw2t = (_Float16*)(ws + FW2T_OFF);
  _Float16* fw3t = (_Float16*)(ws + FW3T_OFF);
  _Float16* z2h  = xh;  // reuse after GEMM1 consumes xh

  const int B = 256;

  // precision conversion: activations f16, weights f16 transposed ([N x K])
  k_f32_to_f16<<<(N_NODES * 128 + B - 1) / B, B, 0, stream>>>(x, xh, N_NODES * 128);
  k_transpose_f16<<<(128 * 128 + B - 1) / B, B, 0, stream>>>(W1, w1t, 128, 128);
  k_transpose_f16<<<(128 * 64 + B - 1) / B, B, 0, stream>>>(W2, w2t, 128, 64);
  k_transpose_f16<<<(128 * 64 + B - 1) / B, B, 0, stream>>>(fw1, fw1t, 128, 64);
  k_transpose_f16<<<(64 * 32 + B - 1) / B, B, 0, stream>>>(fw2, fw2t, 64, 32);
  k_transpose_f16<<<(32 * 16 + B - 1) / B, B, 0, stream>>>(fw3, fw3t, 32, 16);

  // degrees (with self-loops)
  k_deg_init<<<(N_NODES + B - 1) / B, B, 0, stream>>>(deg, N_NODES);
  k_deg_count<<<(N_EDGES + B - 1) / B, B, 0, stream>>>(dst, deg, N_EDGES);
  k_deg_norm<<<(N_NODES + B - 1) / B, B, 0, stream>>>(deg, isd, ind, N_NODES);

  // ---- GCN layer 1: h1 = x @ W1 ; z = relu(agg + h1/deg + b1) ----
  k_zero_f32<<<(N_NODES * 128 + B - 1) / B, B, 0, stream>>>(agg, N_NODES * 128);
  k_gemm<<<(N_NODES / 16) * (128 / 16) / 8, B, 0, stream>>>(xh, w1t, h, N_NODES, 128, 128);
  k_agg<128><<<(int)(((long)N_EDGES * 32 + B - 1) / B), B, 0, stream>>>(src, dst, isd, h, agg, N_EDGES);
  k_finalize<128, true><<<(N_NODES * 128 + B - 1) / B, B, 0, stream>>>(agg, h, ind, b1, zh, N_NODES);

  // ---- GCN layer 2: h2 = z @ W2 ; z2 = agg + h2/deg + b2 ----
  k_zero_f32<<<(N_NODES * 64 + B - 1) / B, B, 0, stream>>>(agg, N_NODES * 64);
  k_gemm<<<(N_NODES / 16) * (64 / 16) / 8, B, 0, stream>>>(zh, w2t, h, N_NODES, 128, 64);
  k_agg<64><<<(int)(((long)N_EDGES * 32 + B - 1) / B), B, 0, stream>>>(src, dst, isd, h, agg, N_EDGES);
  k_finalize<64, false><<<(N_NODES * 64 + B - 1) / B, B, 0, stream>>>(agg, h, ind, b2, z2h, N_NODES);

  // ---- per-edge decoder MLP (WMMA chain) ----
  k_decode<<<(N_EDGES / 16) / 8, B, 0, stream>>>(src, dst, z2h, fw1t, fb1, fw2t, fb2,
                                                 fw3t, fb3, fw4, fb4, out, N_EDGES / 16);
}
